// Torch_Ops_Aten_QuantizedLstm_DataLegacyModule_66236985639428
// MI455X (gfx1250) — compile-verified, matrix operand order checked by reading, not run on hardware
//
#include <hip/hip_runtime.h>

#define T_ 512
#define B_ 64
#define D_ 512
#define H_ 512
#define G_ 2048   // 4*H gates

typedef __attribute__((ext_vector_type(16))) _Float16 v16h;
typedef __attribute__((ext_vector_type(8)))  float    v8f;

union Frag { v16h v; unsigned int u[8]; _Float16 h[16]; };

// ---------------------------------------------------------------- helpers
__global__ void lengths_kernel(const int* __restrict__ bs, int* __restrict__ lens) {
    int b = threadIdx.x;
    if (b < B_) {
        int c = 0;
        for (int t = 0; t < T_; ++t) c += (b < bs[t]) ? 1 : 0;
        lens[b] = c;
    }
}

__global__ void cvt_f32_f16_kernel(const float* __restrict__ in, _Float16* __restrict__ out, long n) {
    long i = (long)blockIdx.x * blockDim.x + threadIdx.x;
    long stride = (long)gridDim.x * blockDim.x;
    for (; i < n; i += stride) out[i] = (_Float16)in[i];
}

// scatter packed rows into dense (T,B,D) f16 buffer (pre-zeroed)
__global__ void pack_kernel(const float* __restrict__ data, const int* __restrict__ pidx,
                            _Float16* __restrict__ x0) {
    int row = blockIdx.x;
    int dst = pidx[row];
    const float* src = data + (long)row * D_;
    _Float16* d = x0 + (long)dst * D_;
    for (int c = threadIdx.x; c < D_; c += blockDim.x) d[c] = (_Float16)src[c];
}

// gather final (T,B,2H) f16 rows back to packed f32 output
__global__ void gather_kernel(const _Float16* __restrict__ xf, const int* __restrict__ pidx,
                              float* __restrict__ out) {
    int row = blockIdx.x;
    int src = pidx[row];
    for (int c = threadIdx.x; c < 2 * H_; c += blockDim.x)
        out[(long)row * (2 * H_) + c] = (float)xf[(long)src * (2 * H_) + c];
}

// ---------------------------------------------------------------- gx = x @ W_ih^T + b (chip-wide WMMA GEMM)
// 4-wide N blocking: one A fragment feeds 4 independent WMMA accumulator chains.
__global__ __launch_bounds__(256)
void gemm_gx_kernel(const _Float16* __restrict__ xin, int din,
                    const _Float16* __restrict__ wih,   // (G, din) row-major f16
                    const float* __restrict__ bias,     // (G)
                    const int* __restrict__ lens, int dirrev,
                    _Float16* __restrict__ gx) {        // (T*B, G) f16, scan-step row order
    const int lane   = threadIdx.x & 31;
    const int gwave  = (blockIdx.x * blockDim.x + threadIdx.x) >> 5;
    const int nwaves = (gridDim.x * blockDim.x) >> 5;
    const int nlo    = lane & 15;
    const int hiSel  = lane >> 4;
    const int khi    = hiSel * 8;
    const int MT = (T_ * B_) / 16;   // 2048 row tiles
    const int NG = G_ / 64;          // 32 groups of 4 col tiles

    for (int it = gwave; it < MT * NG; it += nwaves) {
        int tm = it % MT, ng = it / MT;
        int r = tm * 16 + nlo;               // A lane row = scan-step row (t*B+b)
        int t = r >> 6, b = r & 63;
        int te = t;
        if (dirrev) { te = lens[b] - 1 - t; te = te < 0 ? 0 : te; }
        const _Float16* arow = xin + ((long)te * B_ + b) * din;

        v8f acc[4];
        #pragma unroll
        for (int q = 0; q < 4; ++q) {
            float bv = bias[ng * 64 + q * 16 + nlo];
            acc[q] = (v8f){bv, bv, bv, bv, bv, bv, bv, bv};
        }
        for (int k0 = 0; k0 < din; k0 += 32) {
            Frag a;
            const _Float16* ap = arow + k0 + khi;
            *(uint4*)&a.u[0] = *(const uint4*)ap;
            *(uint4*)&a.u[4] = *(const uint4*)(ap + 16);
            #pragma unroll
            for (int q = 0; q < 4; ++q) {
                Frag w;
                const _Float16* wp = wih + (long)(ng * 64 + q * 16 + nlo) * din + k0 + khi;
                *(uint4*)&w.u[0] = *(const uint4*)wp;
                *(uint4*)&w.u[4] = *(const uint4*)(wp + 16);
                acc[q] = __builtin_amdgcn_wmma_f32_16x16x32_f16(false, a.v, false, w.v,
                                                               (short)0, acc[q], false, false);
            }
        }
        #pragma unroll
        for (int q = 0; q < 4; ++q)
            #pragma unroll
            for (int rr = 0; rr < 8; ++rr) {
                int m = tm * 16 + rr + hiSel * 8;
                gx[(long)m * G_ + ng * 64 + q * 16 + nlo] = (_Float16)acc[q][rr];
            }
    }
}

// ---------------------------------------------------------------- persistent recurrent kernel
// Each wave owns the {i,f,g,o} gate tiles of four 16(batch) x 16(hcol) blocks:
// one LDS A-fragment (h, f16, double-buffered) feeds 4 independent WMMA chains and
// the LSTM cell is evaluated fully in registers (no gates LDS, no cross-wave exchange).
__global__ __launch_bounds__(1024)
void lstm_rec_kernel(const _Float16* __restrict__ gx,   // (T*B, G) f16 scan order
                     const _Float16* __restrict__ whh,  // (G, H) f16
                     float* __restrict__ hst, float* __restrict__ cst,  // (B,H) f32 state
                     const int* __restrict__ lens,
                     _Float16* __restrict__ yout,       // base + dir*H, rowstride 2H
                     int dirrev) {
    const int tid   = threadIdx.x;
    const int lane  = tid & 31;
    const int wave  = tid >> 5;
    const int nlo   = lane & 15;
    const int hiSel = lane >> 4;
    const int khi   = hiSel * 8;

    __shared__ _Float16 hbuf[2][B_ * H_];   // 2 x 64 KB double-buffered h (f16)
    __shared__ int lensLds[B_];

    for (int idx = tid; idx < B_ * H_; idx += 1024) hbuf[0][idx] = (_Float16)hst[idx];
    if (tid < B_) lensLds[tid] = lens[tid];
    __syncthreads();

    for (int t = 0; t < T_; ++t) {
        const _Float16* __restrict__ hcur = hbuf[t & 1];
        _Float16* __restrict__ hnext = hbuf[(t + 1) & 1];
        const _Float16* gxrow = gx + (long)t * B_ * G_;

        // prefetch next timestep's gx block while this step computes
        if (t + 1 < T_) __builtin_prefetch(gxrow + (long)B_ * G_ + (long)tid * 128, 0, 0);

        // 128 (tm,th) combos, 4 per wave; per combo 4 gate tiles (q = i,f,g,o)
        #pragma unroll
        for (int ci = 0; ci < 4; ++ci) {
            const int cb = wave * 4 + ci;
            const int tm = cb >> 5;          // 0..3   (batch tile)
            const int th = cb & 31;          // 0..31  (hcol tile)
            const int hcol = th * 16 + nlo;
            const int bA = tm * 16 + nlo;    // A lane row (batch)

            v8f acc[4];
            #pragma unroll
            for (int q = 0; q < 4; ++q)
                #pragma unroll
                for (int rr = 0; rr < 8; ++rr)
                    acc[q][rr] = (float)gxrow[(long)(tm * 16 + rr + hiSel * 8) * G_ + q * H_ + hcol];

            for (int k0 = 0; k0 < H_; k0 += 32) {
                Frag a;
                const _Float16* ap = hcur + bA * H_ + k0 + khi;
                *(uint4*)&a.u[0] = *(const uint4*)ap;          // ds_load_b128
                *(uint4*)&a.u[4] = *(const uint4*)(ap + 16);
                #pragma unroll
                for (int q = 0; q < 4; ++q) {
                    Frag w;
                    const _Float16* wp = whh + (long)(q * H_ + hcol) * H_ + k0 + khi;
                    *(uint4*)&w.u[0] = *(const uint4*)wp;
                    *(uint4*)&w.u[4] = *(const uint4*)(wp + 16);
                    acc[q] = __builtin_amdgcn_wmma_f32_16x16x32_f16(false, a.v, false, w.v,
                                                                   (short)0, acc[q], false, false);
                }
            }

            // fused LSTM cell for this 16x16 block, fully in registers
            #pragma unroll
            for (int rr = 0; rr < 8; ++rr) {
                int m = tm * 16 + rr + hiSel * 8;    // batch index
                int sidx = m * H_ + hcol;
                bool msk = t < lensLds[m];
                float hold = (float)hcur[sidx];
                float hnv = hold;
                if (msk) {
                    float si = 1.f / (1.f + __expf(-acc[0][rr]));
                    float sf = 1.f / (1.f + __expf(-acc[1][rr]));
                    float tg = tanhf(acc[2][rr]);
                    float so = 1.f / (1.f + __expf(-acc[3][rr]));
                    float cnv = sf * cst[sidx] + si * tg;
                    hnv = so * tanhf(cnv);
                    cst[sidx] = cnv;
                    hst[sidx] = hnv;
                    int tout = dirrev ? (lensLds[m] - 1 - t) : t;
                    yout[((long)tout * B_ + m) * (2 * H_) + hcol] = (_Float16)hnv;
                }
                hnext[sidx] = (_Float16)hnv;   // write to next buffer; no intra-step race
            }
        }
        __syncthreads();   // one barrier per timestep
    }
}

// ---------------------------------------------------------------- launch
extern "C" void kernel_launch(void* const* d_in, const int* in_sizes, int n_in,
                              void* d_out, int out_size, void* d_ws, size_t ws_size,
                              hipStream_t stream) {
    const float* data   = (const float*)d_in[0];
    const float* h0     = (const float*)d_in[1];
    const float* c0     = (const float*)d_in[2];
    const float* w_ih0  = (const float*)d_in[3];
    const float* w_hh0  = (const float*)d_in[4];
    const float* b0     = (const float*)d_in[5];
    const float* w_ih_r = (const float*)d_in[6];
    const float* w_hh_r = (const float*)d_in[7];
    const float* b_r    = (const float*)d_in[8];
    const int*   bsz    = (const int*)d_in[9];
    const int*   pidx   = (const int*)d_in[10];
    float* out = (float*)d_out;
    const int npack = in_sizes[10];

    char* ws = (char*)d_ws;
    size_t off = 0;
    auto alloc = [&](size_t bytes) { char* p = ws + off; off += (bytes + 255) & ~(size_t)255; return p; };
    int*      lens  = (int*)alloc(B_ * sizeof(int));
    _Float16* x0    = (_Float16*)alloc((size_t)T_ * B_ * D_ * 2);        // layer0 input (stride D)
    _Float16* x1    = (_Float16*)alloc((size_t)T_ * B_ * 2 * H_ * 2);    // ping (stride 2H)
    _Float16* x2    = (_Float16*)alloc((size_t)T_ * B_ * 2 * H_ * 2);    // pong
    _Float16* gxbuf = (_Float16*)alloc((size_t)T_ * B_ * G_ * 2);        // per-(l,d) gate preacts
    _Float16* wih0h = (_Float16*)alloc((size_t)2 * G_ * D_ * 2);
    _Float16* whh0h = (_Float16*)alloc((size_t)2 * G_ * H_ * 2);
    _Float16* wihrh = (_Float16*)alloc((size_t)4 * G_ * 2 * H_ * 2);
    _Float16* whhrh = (_Float16*)alloc((size_t)4 * G_ * H_ * 2);

    // h_n / c_n live at the tail of d_out; LSTM updates them in place.
    const size_t stateElems = (size_t)6 * B_ * H_;
    float* hn = out + (size_t)out_size - 2 * stateElems;
    float* cn = out + (size_t)out_size - stateElems;

    hipMemcpyAsync(hn, h0, stateElems * 4, hipMemcpyDeviceToDevice, stream);
    hipMemcpyAsync(cn, c0, stateElems * 4, hipMemcpyDeviceToDevice, stream);
    hipMemsetAsync(x0, 0, (size_t)T_ * B_ * D_ * 2, stream);
    hipMemsetAsync(x1, 0, (size_t)T_ * B_ * 2 * H_ * 2, stream);
    hipMemsetAsync(x2, 0, (size_t)T_ * B_ * 2 * H_ * 2, stream);

    lengths_kernel<<<1, 64, 0, stream>>>(bsz, lens);
    cvt_f32_f16_kernel<<<2048, 256, 0, stream>>>(w_ih0,  wih0h, (long)2 * G_ * D_);
    cvt_f32_f16_kernel<<<2048, 256, 0, stream>>>(w_hh0,  whh0h, (long)2 * G_ * H_);
    cvt_f32_f16_kernel<<<4096, 256, 0, stream>>>(w_ih_r, wihrh, (long)4 * G_ * 2 * H_);
    cvt_f32_f16_kernel<<<2048, 256, 0, stream>>>(w_hh_r, whhrh, (long)4 * G_ * H_);
    pack_kernel<<<npack, 256, 0, stream>>>(data, pidx, x0);

    for (int l = 0; l < 3; ++l) {
        for (int d = 0; d < 2; ++d) {
            const _Float16 *xin, *wih, *whh;
            const float* bias;
            _Float16* yo;
            int din;
            if (l == 0) {
                xin = x0; din = D_; yo = x1;
                wih = wih0h + (size_t)d * G_ * D_;
                whh = whh0h + (size_t)d * G_ * H_;
                bias = b0 + (size_t)d * G_;
            } else {
                xin = (l == 1) ? x1 : x2; din = 2 * H_; yo = (l == 1) ? x2 : x1;
                int wi = (l - 1) * 2 + d;
                wih = wihrh + (size_t)wi * G_ * (2 * H_);
                whh = whhrh + (size_t)wi * G_ * H_;
                bias = b_r + (size_t)wi * G_;
            }
            float* hs = hn + (size_t)(l * 2 + d) * B_ * H_;
            float* cs = cn + (size_t)(l * 2 + d) * B_ * H_;
            gemm_gx_kernel<<<1024, 256, 0, stream>>>(xin, din, wih, bias, lens, d, gxbuf);
            lstm_rec_kernel<<<1, 1024, 0, stream>>>(gxbuf, whh, hs, cs, lens, yo + d * H_, d);
        }
    }
    gather_kernel<<<npack, 256, 0, stream>>>(x1, pidx, out);
}
